// TypedLinear_1236950581828
// MI455X (gfx1250) — matrix operand order repaired and checked by reading, not compiled, not run on hardware
//
#include <hip/hip_runtime.h>

typedef __attribute__((ext_vector_type(16))) _Float16 v16h;
typedef __attribute__((ext_vector_type(8)))  _Float16 v8h;
typedef __attribute__((ext_vector_type(4)))  _Float16 v4h;
typedef __attribute__((ext_vector_type(8)))  float    v8f;

#define NROWS   32768
#define DIN     512
#define DOUT    512
#define NTYPES  8
#define TILE_M  128
#define TILE_N  128
#define TILE_K  32
#define MAX_TILES (NROWS / TILE_M + NTYPES)   // 264
#define LDK     48   // padded halves per LDS row: 32 data + 16 pad, rows stay 16B-aligned (96B)

// ---- workspace layout (int32 units) ----
//  [0..7]                 counts per type
//  [8..15]                scatter cursors
//  [16..23]               segment starts (exclusive scan of counts)
//  [24]                   number of row tiles
//  [32 .. 32+2*MAX_TILES) tile descriptors {type, permStart}
//  [1024 .. 1024+NROWS)   perm (row indices sorted by type)

__device__ __forceinline__ void split2(float x, _Float16& h, _Float16& l) {
    _Float16 hh = (_Float16)x;          // round-to-nearest f16
    h = hh;
    l = (_Float16)(x - (float)hh);      // exact residual in f32, rounded to f16
}

__global__ void tl_init(int* __restrict__ ws) {
    if (threadIdx.x < 32) ws[threadIdx.x] = 0;
}

__global__ void tl_hist(const int* __restrict__ xt, int* __restrict__ ws) {
    __shared__ int lc[NTYPES];
    if (threadIdx.x < NTYPES) lc[threadIdx.x] = 0;
    __syncthreads();
    int i = blockIdx.x * 256 + threadIdx.x;     // grid covers NROWS exactly
    atomicAdd(&lc[xt[i]], 1);
    __syncthreads();
    if (threadIdx.x < NTYPES) atomicAdd(&ws[threadIdx.x], lc[threadIdx.x]);
}

__global__ void tl_scan(int* __restrict__ ws) {
    if (threadIdx.x == 0) {
        int off = 0, nt = 0;
        for (int t = 0; t < NTYPES; ++t) {
            int c = ws[t];
            ws[16 + t] = off;   // segment start
            ws[8 + t]  = off;   // scatter cursor
            for (int s = 0; s < c; s += TILE_M) {
                ws[32 + 2 * nt]     = t;
                ws[32 + 2 * nt + 1] = off + s;
                ++nt;
            }
            off += c;
        }
        ws[24] = nt;
    }
}

__global__ void tl_scatter(const int* __restrict__ xt, int* __restrict__ ws,
                           int* __restrict__ perm) {
    __shared__ int lc[NTYPES], lb[NTYPES];
    if (threadIdx.x < NTYPES) lc[threadIdx.x] = 0;
    __syncthreads();
    int i = blockIdx.x * 256 + threadIdx.x;
    int t = xt[i];
    int p = atomicAdd(&lc[t], 1);               // fast LDS atomic
    __syncthreads();
    if (threadIdx.x < NTYPES)
        lb[threadIdx.x] = atomicAdd(&ws[8 + threadIdx.x], lc[threadIdx.x]);
    __syncthreads();
    perm[lb[t] + p] = i;
}

__global__ __launch_bounds__(256)
void tl_gemm(const float* __restrict__ x, const float* __restrict__ W,
             const int* __restrict__ wsI, const int* __restrict__ perm,
             float* __restrict__ out) {
    const int tile   = blockIdx.x;
    const int nTiles = wsI[24];
    if (tile >= nTiles) return;

    const int t        = wsI[32 + 2 * tile];
    const int rowStart = wsI[32 + 2 * tile + 1];
    const int segEnd   = wsI[16 + t] + wsI[t];
    const int rows     = min(TILE_M, segEnd - rowStart);
    const int nBase    = blockIdx.y * TILE_N;

    const float* Wt = W + (size_t)t * DIN * DOUT;

    const int tid  = threadIdx.x;
    const int lane = tid & 31;
    const int wave = tid >> 5;
    const int wm   = wave & 3;     // M group: rows wm*32..wm*32+31
    const int wn   = wave >> 2;    // N group: cols wn*64..wn*64+63
    const int ln16 = lane & 15;
    const int kh   = lane >> 4;

    // A staging map: row = tid/8 (+p*32), kseg = (tid%8)*4 -> float4, coalesced 128B per 8 lanes
    const int arow0 = tid >> 3;
    const int akseg = (tid & 7) * 4;
    // B staging map: k = tid/32 (+p*8), n = (tid%32)*4 -> float4, coalesced 512B per wave
    const int bk0 = tid >> 5;
    const int bn  = (tid & 31) * 4;

    // Gathered global row index per A-staging pass (-1 => zero-fill)
    int arows[4];
#pragma unroll
    for (int p = 0; p < 4; ++p) {
        int r = arow0 + p * 32;
        arows[p] = (r < rows) ? perm[rowStart + r] : -1;
    }

    __shared__ _Float16 sAh[TILE_M][LDK];
    __shared__ _Float16 sAl[TILE_M][LDK];
    __shared__ _Float16 sBh[TILE_N][LDK];   // B stored transposed: [n][k]
    __shared__ _Float16 sBl[TILE_N][LDK];

    v8f acc[2][4];
#pragma unroll
    for (int mt = 0; mt < 2; ++mt)
#pragma unroll
        for (int nt = 0; nt < 4; ++nt)
            acc[mt][nt] = {};

    for (int kt = 0; kt < DIN; kt += TILE_K) {
        __syncthreads();
        // ---- stage A (gathered rows), split fp32 -> hi/lo f16 ----
#pragma unroll
        for (int p = 0; p < 4; ++p) {
            int r = arow0 + p * 32;
            _Float16 h[4], l[4];
            if (arows[p] >= 0) {
                const float4 v = *(const float4*)(x + (size_t)arows[p] * DIN + kt + akseg);
                split2(v.x, h[0], l[0]); split2(v.y, h[1], l[1]);
                split2(v.z, h[2], l[2]); split2(v.w, h[3], l[3]);
            } else {
#pragma unroll
                for (int i = 0; i < 4; ++i) { h[i] = (_Float16)0.f; l[i] = (_Float16)0.f; }
            }
            v4h hv = { h[0], h[1], h[2], h[3] };
            v4h lv = { l[0], l[1], l[2], l[3] };
            *(v4h*)&sAh[r][akseg] = hv;    // ds_store_b64
            *(v4h*)&sAl[r][akseg] = lv;
        }
        // ---- stage B (single W[t] slice), transpose into [n][k] ----
#pragma unroll
        for (int p = 0; p < 4; ++p) {
            int k = bk0 + p * 8;
            const float4 v = *(const float4*)(Wt + (size_t)(kt + k) * DOUT + nBase + bn);
            _Float16 h, l;
            split2(v.x, h, l); sBh[bn + 0][k] = h; sBl[bn + 0][k] = l;
            split2(v.y, h, l); sBh[bn + 1][k] = h; sBl[bn + 1][k] = l;
            split2(v.z, h, l); sBh[bn + 2][k] = h; sBl[bn + 2][k] = l;
            split2(v.w, h, l); sBh[bn + 3][k] = h; sBl[bn + 3][k] = l;
        }
        __syncthreads();

        // ---- A fragments (ISA 16-bit A 16x32 layout: lane m=lane&15, half kh=lane>>4) ----
        v16h ah[2], al[2];
#pragma unroll
        for (int mt = 0; mt < 2; ++mt) {
            const _Float16* pa = &sAh[wm * 32 + mt * 16 + ln16][kh * 8];
            const _Float16* pl = &sAl[wm * 32 + mt * 16 + ln16][kh * 8];
            v8h a0 = *(const v8h*)pa;          // K = kh*8 .. +7
            v8h a1 = *(const v8h*)(pa + 16);   // K = 16 + kh*8 .. +7
            v8h b0 = *(const v8h*)pl;
            v8h b1 = *(const v8h*)(pl + 16);
            ah[mt] = __builtin_shufflevector(a0, a1, 0,1,2,3,4,5,6,7,8,9,10,11,12,13,14,15);
            al[mt] = __builtin_shufflevector(b0, b1, 0,1,2,3,4,5,6,7,8,9,10,11,12,13,14,15);
        }

        // ---- B fragments + 3-product split-f16 WMMA ----
#pragma unroll
        for (int nt = 0; nt < 4; ++nt) {
            const _Float16* pbh = &sBh[wn * 64 + nt * 16 + ln16][kh * 8];
            const _Float16* pbl = &sBl[wn * 64 + nt * 16 + ln16][kh * 8];
            v16h bh = __builtin_shufflevector(*(const v8h*)pbh, *(const v8h*)(pbh + 16), 0,1,2,3,4,5,6,7,8,9,10,11,12,13,14,15);       // K = kh*8..+7, 16+kh*8..+7 for column n
            v16h bl = __builtin_shufflevector(*(const v8h*)pbl, *(const v8h*)(pbl + 16), 0,1,2,3,4,5,6,7,8,9,10,11,12,13,14,15);
#pragma unroll
            for (int mt = 0; mt < 2; ++mt) {
                acc[mt][nt] = __builtin_amdgcn_wmma_f32_16x16x32_f16(
                    false, ah[mt], false, bh, (short)0, acc[mt][nt], false, false);
                acc[mt][nt] = __builtin_amdgcn_wmma_f32_16x16x32_f16(
                    false, ah[mt], false, bl, (short)0, acc[mt][nt], false, false);
                acc[mt][nt] = __builtin_amdgcn_wmma_f32_16x16x32_f16(
                    false, al[mt], false, bh, (short)0, acc[mt][nt], false, false);
            }
        }
    }

    // ---- scatter C back through perm (f32 C/D layout: row = vgpr + 8*kh, col = lane&15) ----
#pragma unroll
    for (int mt = 0; mt < 2; ++mt) {
#pragma unroll
        for (int v = 0; v < 8; ++v) {
            int r = wm * 32 + mt * 16 + kh * 8 + v;
            if (r < rows) {
                int grow = perm[rowStart + r];
                float* po = out + (size_t)grow * DOUT + nBase + wn * 64 + ln16;
#pragma unroll
                for (int nt = 0; nt < 4; ++nt) po[nt * 16] = acc[mt][nt][v];
            }
        }
    }
}

extern "C" void kernel_launch(void* const* d_in, const int* in_sizes, int n_in,
                              void* d_out, int out_size, void* d_ws, size_t ws_size,
                              hipStream_t stream) {
    const float* x  = (const float*)d_in[0];
    const int*   xt = (const int*)d_in[1];
    const float* W  = (const float*)d_in[2];
    float* out = (float*)d_out;
    int*   ws  = (int*)d_ws;
    int*   perm = ws + 1024;

    tl_init<<<1, 32, 0, stream>>>(ws);
    tl_hist<<<NROWS / 256, 256, 0, stream>>>(xt, ws);
    tl_scan<<<1, 32, 0, stream>>>(ws);
    tl_scatter<<<NROWS / 256, 256, 0, stream>>>(xt, ws, perm);
    tl_gemm<<<dim3(MAX_TILES, DOUT / TILE_N, 1), 256, 0, stream>>>(x, W, ws, perm, out);
}
